// SimpleSSM_627065225419
// MI455X (gfx1250) — matrix operand order repaired and checked by reading, not compiled, not run on hardware
//
#include <hip/hip_runtime.h>
#include <math.h>

// Problem constants (match reference)
#define BSZ_  8
#define LSEQ  2048
#define DDIM  1024
#define MDIM  (BSZ_ * LSEQ)   // 16384

typedef float  v8f   __attribute__((ext_vector_type(8)));
typedef __bf16 v4bf  __attribute__((ext_vector_type(4)));
typedef __bf16 v8bf  __attribute__((ext_vector_type(8)));
typedef __bf16 v16bf __attribute__((ext_vector_type(16)));

// Hardware tanh (V_TANH_F32) if the builtin exists on this toolchain.
#if __has_builtin(__builtin_amdgcn_tanhf)
  #define DEV_TANH(x) __builtin_amdgcn_tanhf(x)
#elif __has_builtin(__builtin_amdgcn_tanh_f32)
  #define DEV_TANH(x) __builtin_amdgcn_tanh_f32(x)
#else
  #define DEV_TANH(x) tanhf(x)
#endif

// K chunk = one bf16 WMMA depth. LDS row stride 40 bf16 = 80 B:
// 16B-aligned rows; 20*i mod 64 banks distinct for i=0..15.
#define KC  32
#define RS  40

// A-fragment (16x32 bf16, MxK): lane(m=l%16, h=l/16):
//   elems 0..7  = K[8h .. 8h+7]       (VGPR0-3)
//   elems 8..15 = K[16+8h .. 16+8h+7] (VGPR4-7)
__device__ __forceinline__ v16bf load_a_frag(const __bf16* plane, int row, int half) {
    const v8bf lo = *(const v8bf*)(plane + row * RS + 8 * half);
    const v8bf hi = *(const v8bf*)(plane + row * RS + 16 + 8 * half);
    return __builtin_shufflevector(lo, hi, 0,1,2,3,4,5,6,7,8,9,10,11,12,13,14,15);
}

// B-fragment (32x16 bf16, KxN), B[k][n] = W[n][k]: lane(n=l%16, h):
//   elems 0..7 = K[8h .. 8h+7], elems 8..15 = K[16+8h .. 16+8h+7], along W's row n.
__device__ __forceinline__ v16bf load_b_frag(const __bf16* plane, int row, int half) {
    const v8bf lo = *(const v8bf*)(plane + row * RS + 8 * half);
    const v8bf hi = *(const v8bf*)(plane + row * RS + 16 + 8 * half);
    return __builtin_shufflevector(lo, hi, 0,1,2,3,4,5,6,7,8,9,10,11,12,13,14,15);
}

// Split one float4 into bf16 hi/lo planes and store 8B to each.
__device__ __forceinline__ void split_store(__bf16* hiP, __bf16* loP, int off, float4 v) {
    v4bf hi = { (__bf16)v.x, (__bf16)v.y, (__bf16)v.z, (__bf16)v.w };
    v4bf lo = { (__bf16)(v.x - (float)hi[0]), (__bf16)(v.y - (float)hi[1]),
                (__bf16)(v.z - (float)hi[2]), (__bf16)(v.w - (float)hi[3]) };
    *(v4bf*)(hiP + off) = hi;
    *(v4bf*)(loP + off) = lo;
}

// out[M,N] = X[M,K] @ W[N,K]^T + bias[N], fp32 in/out, bf16x3 split-precision
// matrix-core inner product (hi*hi + hi*lo + lo*hi, f32 accumulate).
// Block tile 128(M) x 128(N); 8 waves in 4x2; wave tile 32x64 = 2x4 WMMA tiles.
// Double-buffered LDS: one barrier per K chunk.
__global__ __launch_bounds__(256)
void ssm_gemm_xwt_bias(const float* __restrict__ X,
                       const float* __restrict__ W,
                       const float* __restrict__ bias,
                       float* __restrict__ out,
                       int M, int N, int K)
{
    __shared__ __bf16 XsHi[2][128 * RS];
    __shared__ __bf16 XsLo[2][128 * RS];
    __shared__ __bf16 WsHi[2][128 * RS];
    __shared__ __bf16 WsLo[2][128 * RS];

    const int tid  = threadIdx.x;
    const int lane = tid & 31;
    const int wave = tid >> 5;
    const int half = lane >> 4;
    const int l16  = lane & 15;

    const int block_m = blockIdx.x * 128;
    const int block_n = blockIdx.y * 128;
    const int wm = (wave & 3) * 32;   // wave M offset within block tile
    const int wn = (wave >> 2) * 64;  // wave N offset within block tile

    // Staging addressing: 128 rows x 32 cols = 1024 float4 per tile; 4/thread.
    const int sr  = tid >> 3;       // base row 0..31 (it adds +32 each)
    const int sc4 = tid & 7;        // float4 column 0..7

    v8f acc[2][4];
#pragma unroll
    for (int i = 0; i < 2; ++i)
#pragma unroll
        for (int j = 0; j < 4; ++j) {
            v8f z = {0.f, 0.f, 0.f, 0.f, 0.f, 0.f, 0.f, 0.f};
            acc[i][j] = z;
        }

    float4 xs[4], ws[4];

    // ---- prologue: fetch + split chunk 0 into buffer 0 ----
#pragma unroll
    for (int it = 0; it < 4; ++it) {
        int r = sr + it * 32;
        xs[it] = *(const float4*)(X + (size_t)(block_m + r) * K + sc4 * 4);
        ws[it] = *(const float4*)(W + (size_t)(block_n + r) * K + sc4 * 4);
    }
#pragma unroll
    for (int it = 0; it < 4; ++it) {
        int r = sr + it * 32;
        split_store(XsHi[0], XsLo[0], r * RS + sc4 * 4, xs[it]);
        split_store(WsHi[0], WsLo[0], r * RS + sc4 * 4, ws[it]);
    }
    __syncthreads();

    const int NC = K / KC;
    for (int c = 0; c < NC; ++c) {
        const int buf = c & 1;

        // issue next chunk's global loads early (latency hidden under WMMA)
        if (c + 1 < NC) {
            const int k0 = (c + 1) * KC;
#pragma unroll
            for (int it = 0; it < 4; ++it) {
                int r = sr + it * 32;
                xs[it] = *(const float4*)(X + (size_t)(block_m + r) * K + k0 + sc4 * 4);
                ws[it] = *(const float4*)(W + (size_t)(block_n + r) * K + k0 + sc4 * 4);
            }
        }

        // ---- bf16x3 WMMA: 2x4 subtiles x 3 products per K=32 chunk ----
        v16bf ah[2], al[2], bh[4], bl[4];
#pragma unroll
        for (int i = 0; i < 2; ++i) {
            ah[i] = load_a_frag(XsHi[buf], wm + i * 16 + l16, half);
            al[i] = load_a_frag(XsLo[buf], wm + i * 16 + l16, half);
        }
#pragma unroll
        for (int j = 0; j < 4; ++j) {
            bh[j] = load_b_frag(WsHi[buf], wn + j * 16 + l16, half);
            bl[j] = load_b_frag(WsLo[buf], wn + j * 16 + l16, half);
        }
#pragma unroll
        for (int i = 0; i < 2; ++i)
#pragma unroll
            for (int j = 0; j < 4; ++j) {
                acc[i][j] = __builtin_amdgcn_wmma_f32_16x16x32_bf16(
                    false, ah[i], false, bh[j], (short)0, acc[i][j], false, false);
                acc[i][j] = __builtin_amdgcn_wmma_f32_16x16x32_bf16(
                    false, ah[i], false, bl[j], (short)0, acc[i][j], false, false);
                acc[i][j] = __builtin_amdgcn_wmma_f32_16x16x32_bf16(
                    false, al[i], false, bh[j], (short)0, acc[i][j], false, false);
            }

        // split + store next chunk into the other buffer (no conflict with readers)
        if (c + 1 < NC) {
#pragma unroll
            for (int it = 0; it < 4; ++it) {
                int r = sr + it * 32;
                split_store(XsHi[buf ^ 1], XsLo[buf ^ 1], r * RS + sc4 * 4, xs[it]);
                split_store(WsHi[buf ^ 1], WsLo[buf ^ 1], r * RS + sc4 * 4, ws[it]);
            }
        }
        __syncthreads();
    }

    // ---- epilogue: f32 C/D layout: VGPR r, lane l -> (M = r + 8*half, N = l16) ----
#pragma unroll
    for (int i = 0; i < 2; ++i) {
#pragma unroll
        for (int j = 0; j < 4; ++j) {
            int gn = block_n + wn + j * 16 + l16;
            float bb = bias[gn];
#pragma unroll
            for (int r = 0; r < 8; ++r) {
                int gm = block_m + wm + i * 16 + r + 8 * half;
                out[(size_t)gm * N + gn] = acc[i][j][r] + bb;
            }
        }
    }
}

// In-place sequential scan: h_t = tanh(A ⊙ h_{t-1} + Bx_t); overwrites Bx with h.
// One thread per (b, d/4); coalesced float4 streaming along L.
// Next step's load is rotated ahead of the dependent fma+tanh chain.
__global__ __launch_bounds__(64)
void ssm_scan(float* __restrict__ BxH, const float* __restrict__ A)
{
    const int idx = blockIdx.x * 64 + threadIdx.x;   // 0..2047
    const int dv  = idx & (DDIM / 4 - 1);            // float4 index in D
    const int b   = idx >> 8;                        // idx / (D/4)

    const float4 a4 = *(const float4*)(A + dv * 4);
    float4 h = {0.f, 0.f, 0.f, 0.f};
    float* p = BxH + (size_t)b * LSEQ * DDIM + dv * 4;

    float4 bx = *(const float4*)p;
    for (int l = 0; l < LSEQ - 1; ++l) {
        float4 nxt = *(const float4*)(p + DDIM);     // independent of tanh chain
        h.x = DEV_TANH(fmaf(a4.x, h.x, bx.x));
        h.y = DEV_TANH(fmaf(a4.y, h.y, bx.y));
        h.z = DEV_TANH(fmaf(a4.z, h.z, bx.z));
        h.w = DEV_TANH(fmaf(a4.w, h.w, bx.w));
        *(float4*)p = h;
        p += DDIM;
        bx = nxt;
    }
    h.x = DEV_TANH(fmaf(a4.x, h.x, bx.x));
    h.y = DEV_TANH(fmaf(a4.y, h.y, bx.y));
    h.z = DEV_TANH(fmaf(a4.z, h.z, bx.z));
    h.w = DEV_TANH(fmaf(a4.w, h.w, bx.w));
    *(float4*)p = h;
}

extern "C" void kernel_launch(void* const* d_in, const int* in_sizes, int n_in,
                              void* d_out, int out_size, void* d_ws, size_t ws_size,
                              hipStream_t stream) {
    const float* x   = (const float*)d_in[0];
    const float* A   = (const float*)d_in[1];
    const float* W_B = (const float*)d_in[2];
    const float* b_B = (const float*)d_in[3];
    const float* W_C = (const float*)d_in[4];
    const float* b_C = (const float*)d_in[5];
    float* y  = (float*)d_out;
    float* Bx = (float*)d_ws;   // [B, L, D] fp32 = 64 MB scratch, reused in place for h

    dim3 grid(MDIM / 128, DDIM / 128);   // (128, 8)

    // Phase 1: Bx = x @ W_B^T + b_B
    ssm_gemm_xwt_bias<<<grid, 256, 0, stream>>>(x, W_B, b_B, Bx, MDIM, DDIM, DDIM);
    // Phase 2: sequential tanh recurrence, in place (Bx -> h)
    ssm_scan<<<(BSZ_ * DDIM / 4) / 64, 64, 0, stream>>>(Bx, A);
    // Phase 3: y = h @ W_C^T + b_C
    ssm_gemm_xwt_bias<<<grid, 256, 0, stream>>>(Bx, W_C, b_C, y, MDIM, DDIM, DDIM);
}
